// AttentionPooling_68599217652459
// MI455X (gfx1250) — compile-verified, matrix operand order checked by reading, not run on hardware
//
#include <hip/hip_runtime.h>
#include <hip/hip_bf16.h>

// Problem constants (from reference)
constexpr int B = 32, S = 4096, E = 1024, H = 16, D = 64;
constexpr float SCALE = 0.125f;                 // 1/sqrt(D), D=64

typedef __attribute__((ext_vector_type(16))) __bf16 v16bf;
typedef __attribute__((ext_vector_type(8)))  __bf16 v8bf;
typedef __attribute__((ext_vector_type(8)))  float  v8f;
typedef __attribute__((ext_vector_type(4)))  float  vf4;   // clang ext-vector (nontemporal-compatible)

union BF16x16 {
  v16bf v;
  v8bf  h[2];
  unsigned short u[16];
  uint4 q[2];
};

union BF16x8 {
  v8bf v;
  unsigned short u[8];
};

// Packed f32 -> bf16 (vector fptrunc => v_cvt_pk_bf16_f32, hardware RNE)
__device__ __forceinline__ v8bf cvt8f(const vf4 f0, const vf4 f1) {
  v8f f = {f0.x, f0.y, f0.z, f0.w, f1.x, f1.y, f1.z, f1.w};
  return __builtin_convertvector(f, v8bf);
}

__device__ __forceinline__ unsigned short f2bf1(float f) {
  union { __bf16 b; unsigned short u; } c;
  c.b = (__bf16)f;
  return c.u;
}

// ---------------------------------------------------------------------------
// K0a: q[b,i] = Wq[i,:] . x[b,0,:] + bq[i]           (B*E outputs, 1024 MACs)
// ---------------------------------------------------------------------------
__global__ __launch_bounds__(256) void k_qproj(const float* __restrict__ x,
                                               const float* __restrict__ w,
                                               const float* __restrict__ bias,
                                               float* __restrict__ q) {
  int idx = blockIdx.x * 256 + threadIdx.x;     // b*E + i
  int b = idx >> 10, i = idx & (E - 1);
  const float* wr = w + (size_t)i * E;
  const float* xr = x + (size_t)b * S * E;      // x[b,0,:]
  float acc = bias[i];
  for (int e = 0; e < E; e += 4) {
    vf4 wv = *(const vf4*)(wr + e);
    vf4 xv = *(const vf4*)(xr + e);
    acc += wv.x * xv.x + wv.y * xv.y + wv.z * xv.z + wv.w * xv.w;
  }
  q[idx] = acc;
}

// ---------------------------------------------------------------------------
// K0b: r_bf[b,h,e] = (1/sqrt(D)) * sum_d Wk[h*D+d, e] * q[b,h*D+d]   (bf16)
// ---------------------------------------------------------------------------
__global__ __launch_bounds__(256) void k_rproj(const float* __restrict__ wk,   // = in_proj_w + E*E
                                               const float* __restrict__ q,
                                               unsigned short* __restrict__ rbf) {
  int idx = blockIdx.x * 256 + threadIdx.x;     // (b*H+h)*E + e
  int e  = idx & (E - 1);
  int bh = idx >> 10;
  int h  = bh & (H - 1);
  int b  = bh >> 4;
  const float* qp   = q  + (size_t)b * E + h * D;
  const float* wcol = wk + (size_t)(h * D) * E + e;
  float acc = 0.f;
  for (int d = 0; d < D; ++d) acc += wcol[(size_t)d * E] * qp[d];
  rbf[idx] = f2bf1(acc * SCALE);
}

// ---------------------------------------------------------------------------
// K1: scores[b] (16 x S) = R[b] (16 x E, bf16) @ x[b]^T     [WMMA, K=E]
// One wave = one 16x16 score tile; 8 waves/block; grid = B * S/128.
// x is streamed non-temporally (512 MB, no reuse within the pass).
// ---------------------------------------------------------------------------
__global__ __launch_bounds__(256) void k_scores(const float* __restrict__ x,
                                                const unsigned short* __restrict__ rbf,
                                                float* __restrict__ scores) {
  const int wave = threadIdx.x >> 5;
  const int lane = threadIdx.x & 31;
  const int b    = blockIdx.x >> 5;             // 32 s-groups per b
  const int sg   = blockIdx.x & 31;
  const int s0   = (sg * 8 + wave) * 16;
  const int half = lane >> 4;                   // ISA lane-layout half
  const int row  = lane & 15;                   // A: M index; B: N index
  const int koff = half * 8;                    // K sub-chunk per half

  const unsigned short* arow = rbf + (size_t)(b * H + row) * E;
  const float*          xrow = x   + ((size_t)b * S + (s0 + row)) * E;

  v8f c = {};
#pragma unroll 2
  for (int k0 = 0; k0 < E; k0 += 32) {
    BF16x16 a, bm;
    a.q[0] = *(const uint4*)(arow + k0 + koff);        // K = koff..koff+7
    a.q[1] = *(const uint4*)(arow + k0 + 16 + koff);   // K = 16+koff..
    const vf4* xp0 = (const vf4*)(xrow + k0 + koff);
    const vf4* xp1 = (const vf4*)(xrow + k0 + 16 + koff);
    vf4 f0 = __builtin_nontemporal_load(xp0);
    vf4 f1 = __builtin_nontemporal_load(xp0 + 1);
    vf4 f2 = __builtin_nontemporal_load(xp1);
    vf4 f3 = __builtin_nontemporal_load(xp1 + 1);
    bm.h[0] = cvt8f(f0, f1);
    bm.h[1] = cvt8f(f2, f3);
    c = __builtin_amdgcn_wmma_f32_16x16x32_bf16(false, a.v, false, bm.v,
                                                (short)0, c, false, false);
  }
  const int mbase = half * 8;                   // C layout: lanes>=16 hold M=8..15
  float* srow = scores + ((size_t)(b * H + mbase)) * S + s0 + row;
#pragma unroll
  for (int i = 0; i < 8; ++i) srow[(size_t)i * S] = c[i];
}

// ---------------------------------------------------------------------------
// K2: masked softmax over S, in place.  One block per (b,h).
// ---------------------------------------------------------------------------
__global__ __launch_bounds__(256) void k_softmax(float* __restrict__ scores,
                                                 const unsigned char* __restrict__ mask) {
  __shared__ float red[256];
  const int bh = blockIdx.x;
  const int b  = bh >> 4;
  float* row = scores + (size_t)bh * S;
  const unsigned char* mrow = mask + (size_t)b * S;
  const int t = threadIdx.x;

  float vals[16];
  float mx = -INFINITY;
#pragma unroll
  for (int j = 0; j < 16; ++j) {
    int s = t + j * 256;
    float v = mrow[s] ? -1e30f : row[s];
    vals[j] = v;
    mx = fmaxf(mx, v);
  }
  red[t] = mx; __syncthreads();
  for (int off = 128; off > 0; off >>= 1) {
    if (t < off) red[t] = fmaxf(red[t], red[t + off]);
    __syncthreads();
  }
  mx = red[0]; __syncthreads();

  float sum = 0.f;
#pragma unroll
  for (int j = 0; j < 16; ++j) {
    float e = __expf(vals[j] - mx);
    vals[j] = e; sum += e;
  }
  red[t] = sum; __syncthreads();
  for (int off = 128; off > 0; off >>= 1) {
    if (t < off) red[t] += red[t + off];
    __syncthreads();
  }
  float inv = 1.0f / red[0];
#pragma unroll
  for (int j = 0; j < 16; ++j) row[j * 256 + t] = vals[j] * inv;
}

// ---------------------------------------------------------------------------
// K2b: attn_weights[b,s] = mean_h attn[b,h,s]   -> d_out[B*E + b*S + s]
// ---------------------------------------------------------------------------
__global__ __launch_bounds__(256) void k_attnw(const float* __restrict__ attn,
                                               float* __restrict__ outw) {
  int idx = blockIdx.x * 256 + threadIdx.x;     // b*S + s
  int b = idx >> 12, s = idx & (S - 1);
  float acc = 0.f;
#pragma unroll
  for (int h = 0; h < H; ++h) acc += attn[((size_t)(b * H + h)) * S + s];
  outw[idx] = acc * (1.0f / H);
}

// ---------------------------------------------------------------------------
// K3: p[b] (16 x E) = attn[b] (16 x S) @ x[b] (S x E)       [WMMA, K=S]
// Block = one b, one 128-wide e-group; x tile transposed through LDS.
// Each thread owns an (s, s+1) pair x 8 e-values so adjacent-s bf16 pack
// into dword LDS stores (8 x ds_store_b32 instead of 16 x ds_store_b16).
// ---------------------------------------------------------------------------
constexpr int K3P = 40;                          // LDS pitch (ushorts), keeps 16B align
__global__ __launch_bounds__(256) void k_pool(const float* __restrict__ x,
                                              const float* __restrict__ attn,
                                              float* __restrict__ p) {
  __shared__ unsigned short lds[128 * K3P];      // x^T tile: [e_local][k], bf16
  const int wave = threadIdx.x >> 5;
  const int lane = threadIdx.x & 31;
  const int b  = blockIdx.x >> 3;                // 8 e-groups per b
  const int e0 = (blockIdx.x & 7) * 128;
  const int half = lane >> 4;
  const int row  = lane & 15;
  const int koff = half * 8;
  const int sp  = (threadIdx.x & 15) * 2;        // s pair base: 0,2,..,30
  const int blk = threadIdx.x >> 4;              // 0..15 : 8-wide e block

  v8f c = {};
  for (int k0 = 0; k0 < S; k0 += 32) {
    // cooperative load x[b, k0+sp(+1), e0+blk*8 .. +7], transpose to LDS bf16
    const float* xr0 = x + ((size_t)b * S + (k0 + sp)) * E + e0 + blk * 8;
    vf4 r00 = __builtin_nontemporal_load((const vf4*)xr0);
    vf4 r01 = __builtin_nontemporal_load((const vf4*)xr0 + 1);
    vf4 r10 = __builtin_nontemporal_load((const vf4*)(xr0 + E));
    vf4 r11 = __builtin_nontemporal_load((const vf4*)(xr0 + E) + 1);
    BF16x8 c0, c1;
    c0.v = cvt8f(r00, r01);
    c1.v = cvt8f(r10, r11);
#pragma unroll
    for (int j = 0; j < 8; ++j) {
      unsigned int pk = (unsigned int)c0.u[j] | ((unsigned int)c1.u[j] << 16);
      *(unsigned int*)&lds[(blk * 8 + j) * K3P + sp] = pk;
    }
    __syncthreads();

    BF16x16 a, bm;
    const float* ar = attn + (size_t)(b * H + row) * S + k0;
    a.h[0] = cvt8f(*(const vf4*)(ar + koff),      *(const vf4*)(ar + koff + 4));
    a.h[1] = cvt8f(*(const vf4*)(ar + 16 + koff), *(const vf4*)(ar + 16 + koff + 4));
    const int nl = wave * 16 + row;
    bm.q[0] = *(const uint4*)(&lds[nl * K3P + koff]);
    bm.q[1] = *(const uint4*)(&lds[nl * K3P + koff + 16]);
    c = __builtin_amdgcn_wmma_f32_16x16x32_bf16(false, a.v, false, bm.v,
                                                (short)0, c, false, false);
    __syncthreads();
  }
  const int mbase = half * 8;
  float* pr = p + (size_t)(b * H + mbase) * E + e0 + wave * 16 + row;
#pragma unroll
  for (int i = 0; i < 8; ++i) pr[(size_t)i * E] = c[i];
}

// ---------------------------------------------------------------------------
// K4: ctx[b, h*D+d] = bv[i] + Wv[i,:] . p[b,h,:]
// ---------------------------------------------------------------------------
__global__ __launch_bounds__(256) void k_ctx(const float* __restrict__ wv,   // in_proj_w + 2*E*E
                                             const float* __restrict__ bv,   // in_proj_b + 2*E
                                             const float* __restrict__ p,
                                             float* __restrict__ ctx) {
  int idx = blockIdx.x * 256 + threadIdx.x;     // b*E + i
  int b = idx >> 10, i = idx & (E - 1);
  int h = i >> 6;                               // D = 64
  const float* wr = wv + (size_t)i * E;
  const float* pr = p + (size_t)(b * H + h) * E;
  float acc = bv[i];
  for (int e = 0; e < E; e += 4) {
    vf4 wvv = *(const vf4*)(wr + e);
    vf4 pv  = *(const vf4*)(pr + e);
    acc += wvv.x * pv.x + wvv.y * pv.y + wvv.z * pv.z + wvv.w * pv.w;
  }
  ctx[idx] = acc;
}

// ---------------------------------------------------------------------------
// K5: out[b,i] = out_b[i] + out_w[i,:] . ctx[b,:]
// ---------------------------------------------------------------------------
__global__ __launch_bounds__(256) void k_out(const float* __restrict__ ow,
                                             const float* __restrict__ ob,
                                             const float* __restrict__ ctx,
                                             float* __restrict__ out) {
  int idx = blockIdx.x * 256 + threadIdx.x;     // b*E + i
  int b = idx >> 10, i = idx & (E - 1);
  const float* wr = ow + (size_t)i * E;
  const float* cr = ctx + (size_t)b * E;
  float acc = ob[i];
  for (int e = 0; e < E; e += 4) {
    vf4 wvv = *(const vf4*)(wr + e);
    vf4 cv  = *(const vf4*)(cr + e);
    acc += wvv.x * cv.x + wvv.y * cv.y + wvv.z * cv.z + wvv.w * cv.w;
  }
  out[idx] = acc;
}

// ---------------------------------------------------------------------------
// Workspace layout (bytes): q | r_bf | scores/attn | p | ctx  (~11.3 MB)
// ---------------------------------------------------------------------------
extern "C" void kernel_launch(void* const* d_in, const int* in_sizes, int n_in,
                              void* d_out, int out_size, void* d_ws, size_t ws_size,
                              hipStream_t stream) {
  const float*         x     = (const float*)d_in[0];
  const unsigned char* mask  = (const unsigned char*)d_in[1];  // jax bool = 1 byte
  const float*         in_w  = (const float*)d_in[2];          // (3E, E)
  const float*         in_b  = (const float*)d_in[3];          // (3E,)
  const float*         out_w = (const float*)d_in[4];          // (E, E)
  const float*         out_b = (const float*)d_in[5];          // (E,)
  float* out = (float*)d_out;                                  // out (B*E) ++ attn_w (B*S)

  char* ws = (char*)d_ws;
  size_t off = 0;
  float*          q      = (float*)(ws + off);          off += (size_t)B * E * 4;
  unsigned short* rbf    = (unsigned short*)(ws + off); off += (size_t)B * H * E * 2;
  float*          scores = (float*)(ws + off);          off += (size_t)B * H * S * 4;
  float*          p      = (float*)(ws + off);          off += (size_t)B * H * E * 4;
  float*          ctx    = (float*)(ws + off);          off += (size_t)B * E * 4;

  k_qproj  <<<(B * E) / 256,      256, 0, stream>>>(x, in_w, in_b, q);
  k_rproj  <<<(B * H * E) / 256,  256, 0, stream>>>(in_w + (size_t)E * E, q, rbf);
  k_scores <<<B * (S / 128),      256, 0, stream>>>(x, rbf, scores);
  k_softmax<<<B * H,              256, 0, stream>>>(scores, mask);
  k_attnw  <<<(B * S) / 256,      256, 0, stream>>>(scores, out + (size_t)B * E);
  k_pool   <<<B * (E / 128),      256, 0, stream>>>(x, scores, p);
  k_ctx    <<<(B * E) / 256,      256, 0, stream>>>(in_w + 2 * (size_t)E * E,
                                                    in_b + 2 * E, p, ctx);
  k_out    <<<(B * E) / 256,      256, 0, stream>>>(out_w, out_b, ctx, out);
}
